// model2_34918084116710
// MI455X (gfx1250) — matrix operand order repaired and checked by reading, not compiled, not run on hardware
//
#include <hip/hip_runtime.h>
#include <hip/hip_bf16.h>
#include <stdint.h>

// ---------------------------------------------------------------------------
// GCN + SAGPool pipeline for MI455X (gfx1250, wave32, WMMA)
// Dense GEMMs: bf16 WMMA 16x16x32, f32 accumulate. Each wave computes a
// 16x64 output strip (4 WMMAs per A-fragment); the 64-column W^T panel is
// staged in LDS once per block via the Tensor Data Mover (TDM) when the
// builtin exists, else via cooperative vector loads.
// Scatter/gather stages: f32 global atomics. Top-k: per-graph radix select
// using LDS histograms. Readout: wave32 shuffle reductions.
// ---------------------------------------------------------------------------

typedef __attribute__((ext_vector_type(16))) __bf16   v16bf;
typedef __attribute__((ext_vector_type(8)))  __bf16   v8bf;
typedef __attribute__((ext_vector_type(8)))  float    v8f;
typedef __attribute__((ext_vector_type(4)))  unsigned u32x4;
typedef __attribute__((ext_vector_type(8)))  int      i32x8;
typedef __attribute__((ext_vector_type(4)))  int      i32x4;

#define BGRAPH 8
#define EDGES  384000

#if defined(__HIP_DEVICE_COMPILE__) && __has_builtin(__builtin_amdgcn_tensor_load_to_lds) && __has_builtin(__builtin_amdgcn_s_wait_tensorcnt)
#define HAVE_TDM 1
#else
#define HAVE_TDM 0
#endif

// ---- small helpers ---------------------------------------------------------

__device__ __forceinline__ unsigned f2ord(float f) {
    unsigned u = __float_as_uint(f);
    return u ^ ((u & 0x80000000u) ? 0xFFFFFFFFu : 0x80000000u);
}

// ---- init kernels ----------------------------------------------------------

__global__ void k_init_edges(const int* __restrict__ ei, int* row, int* col,
                             float* w, int E) {
    int e = blockIdx.x * blockDim.x + threadIdx.x;
    if (e < E) { row[e] = ei[e]; col[e] = ei[E + e]; w[e] = 1.0f; }
}

// W row-major [K][N] f32  ->  Wt row-major [N][K] bf16 (so each WMMA B-lane
// reads two contiguous 8-element K-runs of one output column)
__global__ void k_wt_bf16(const float* __restrict__ W, __bf16* __restrict__ Wt,
                          int K, int N) {
    int t = blockIdx.x * blockDim.x + threadIdx.x;
    if (t < K * N) {
        int n = t / K, k = t % K;
        Wt[(long)n * K + k] = (__bf16)W[(long)k * N + n];
    }
}

__global__ void k_bias_init(float* __restrict__ out, const float* __restrict__ bias,
                            long total, int C) {
    long t = (long)blockIdx.x * blockDim.x + threadIdx.x;
    if (t < total) out[t] = bias[(int)(t % C)];
}

// ---- WMMA GEMM: D[M,N] = A[M,K](f32 -> bf16) * Wt[N,K]^T -------------------
// block = 128 threads = 4 waves. Block tile = 64x64.
// Wave w owns M-rows [blockIdx.x*64 + w*16, +16) and all 4 N-subtiles.
// B panel (64 columns x K bf16) staged in LDS: TDM if available.
__global__ void k_gemm_bf16(const float* __restrict__ A,
                            const __bf16* __restrict__ Wt,
                            float* __restrict__ D, int M, int K, int N) {
    extern __shared__ __bf16 sB[];                 // [64][K]
    const int wave = threadIdx.x >> 5;
    const int lane = threadIdx.x & 31;
    const int l16  = lane & 15;
    const int half = lane >> 4;                    // 0 or 1
    const long m0   = (long)blockIdx.x * 64 + wave * 16;
    const long nblk = (long)blockIdx.y * 64;

#if HAVE_TDM
    // --- Tensor Data Mover: 2D tile, 64 rows of K bf16, row stride K ---
    if (wave == 0) {
        unsigned lds = (unsigned)(uintptr_t)sB;    // LDS aperture: addr[31:0]
        unsigned long long ga = (unsigned long long)(uintptr_t)(Wt + nblk * K);
        u32x4 g0;
        g0[0] = 1u;                                          // count=1, user D#
        g0[1] = lds;                                         // lds_addr
        g0[2] = (unsigned)(ga & 0xFFFFFFFFu);                // global_addr[31:0]
        g0[3] = (unsigned)((ga >> 32) & 0x1FFFFFFu) | (2u << 30); // [56:32]|type=2
        i32x8 g1;
        g1[0] = 1 << 16;                                     // data_size=2B
        g1[1] = (K & 0xFFFF) << 16;                          // tensor_dim0 lo16
        g1[2] = ((K >> 16) & 0xFFFF) | (64 << 16);           // dim0 hi | tensor_dim1
        g1[3] = (K & 0xFFFF) << 16;                          // tile_dim0 = K
        g1[4] = 64;                                          // tile_dim1 = 64 rows
        g1[5] = K;                                           // tensor_dim0_stride
        g1[6] = 0;
        g1[7] = 0;
        i32x4 z4 = {0, 0, 0, 0};
#if __clang_major__ >= 23
        i32x8 z8 = {0, 0, 0, 0, 0, 0, 0, 0};
        __builtin_amdgcn_tensor_load_to_lds(g0, g1, z4, z4, z8, 0);
#else
        __builtin_amdgcn_tensor_load_to_lds(g0, g1, z4, z4, 0);
#endif
        __builtin_amdgcn_s_wait_tensorcnt(0);
    }
    __syncthreads();
#else
    {   // cooperative fallback: copy 64*K bf16 panel to LDS
        const float4* src = (const float4*)(Wt + nblk * K);
        float4* dst = (float4*)sB;
        int n16 = (64 * K) >> 3;                   // 16B chunks
        for (int i = threadIdx.x; i < n16; i += (int)blockDim.x) dst[i] = src[i];
        __syncthreads();
    }
#endif

    const float* arow = A + (m0 + l16) * (long)K;

    v8f acc[4];
#pragma unroll
    for (int t = 0; t < 4; ++t) acc[t] = (v8f){0.f,0.f,0.f,0.f,0.f,0.f,0.f,0.f};

    for (int k = 0; k < K; k += 32) {
        __builtin_prefetch(arow + k + 256, 0, 1);  // global_prefetch_b8
        // A fragment: lanes 0-15 hold K {k..k+7, k+16..k+23} of row m0+l16,
        // lanes 16-31 hold K {k+8..k+15, k+24..k+31}.
        const float4 a0 = *(const float4*)(arow + k + half * 8);
        const float4 a1 = *(const float4*)(arow + k + half * 8 + 4);
        const float4 a2 = *(const float4*)(arow + k + 16 + half * 8);
        const float4 a3 = *(const float4*)(arow + k + 16 + half * 8 + 4);
        v16bf a;
        a[0]  = (__bf16)a0.x; a[1]  = (__bf16)a0.y; a[2]  = (__bf16)a0.z; a[3]  = (__bf16)a0.w;
        a[4]  = (__bf16)a1.x; a[5]  = (__bf16)a1.y; a[6]  = (__bf16)a1.z; a[7]  = (__bf16)a1.w;
        a[8]  = (__bf16)a2.x; a[9]  = (__bf16)a2.y; a[10] = (__bf16)a2.z; a[11] = (__bf16)a2.w;
        a[12] = (__bf16)a3.x; a[13] = (__bf16)a3.y; a[14] = (__bf16)a3.z; a[15] = (__bf16)a3.w;
#pragma unroll
        for (int t = 0; t < 4; ++t) {
            // B fragment from LDS: column nblk + t*16 + l16, K {k+half*8,+8} and {k+16+half*8,+8}
            const __bf16* bp = sB + ((size_t)(t * 16 + l16) * K + k + half * 8);
            const v8bf blo = *(const v8bf*)(bp);
            const v8bf bhi = *(const v8bf*)(bp + 16);
            v16bf b;
#pragma unroll
            for (int i = 0; i < 8; ++i) { b[i] = blo[i]; b[i + 8] = bhi[i]; }
            acc[t] = __builtin_amdgcn_wmma_f32_16x16x32_bf16(
                         false, a, false, b, (short)0, acc[t], false, false);
        }
    }

    const long mbase = m0 + half * 8;              // VGPR v -> M = v + 8*half
#pragma unroll
    for (int t = 0; t < 4; ++t)
#pragma unroll
        for (int v = 0; v < 8; ++v)
            D[(mbase + v) * (long)N + nblk + t * 16 + l16] = acc[t][v];
}

// ---- degree / normalization ------------------------------------------------

__global__ void k_deg(const int* __restrict__ col, const float* __restrict__ w,
                      float* deg, int E) {
    int e = blockIdx.x * blockDim.x + threadIdx.x;
    if (e < E) atomicAdd(&deg[col[e]], w[e]);
}

__global__ void k_dinv(const float* __restrict__ deg, float* dinv, int n) {
    int i = blockIdx.x * blockDim.x + threadIdx.x;
    if (i < n) {
        float d = deg[i];
        dinv[i] = d > 0.f ? rsqrtf(fmaxf(d, 1e-12f)) : 0.f;
    }
}

// ---- edge scatter: out[col] += norm * h[row]  (dinv==nullptr -> norm = w) --
__global__ void k_scatter(const float* __restrict__ h, const int* __restrict__ row,
                          const int* __restrict__ col, const float* __restrict__ w,
                          const float* __restrict__ dinv, float* __restrict__ out,
                          int E, int C) {
    long t = (long)blockIdx.x * blockDim.x + threadIdx.x;
    int per = C >> 2;
    long e = t / per;
    if (e >= E) return;
    float wt = w[e];
    if (wt == 0.f) return;
    int c = (int)(t % per) << 2;
    int r = row[e], cl = col[e];
    float nm = dinv ? dinv[r] * wt * dinv[cl] : wt;
    const float4 hv = *(const float4*)(h + (long)r * C + c);
    float* o = out + (long)cl * C + c;
    atomicAdd(o + 0, nm * hv.x);
    atomicAdd(o + 1, nm * hv.y);
    atomicAdd(o + 2, nm * hv.z);
    atomicAdd(o + 3, nm * hv.w);
}

// ---- batch norm (biased batch stats) + ReLU --------------------------------

__global__ void k_colstats(const float* __restrict__ h, long n, int C, int rowsPer,
                           float* __restrict__ sum, float* __restrict__ sumsq) {
    int c = blockIdx.x * blockDim.x + threadIdx.x;
    if (c >= C) return;
    long r0 = (long)blockIdx.y * rowsPer;
    long r1 = r0 + rowsPer; if (r1 > n) r1 = n;
    float s = 0.f, q = 0.f;
    for (long r = r0; r < r1; ++r) {
        float v = h[r * C + c];
        s += v; q += v * v;
    }
    atomicAdd(&sum[c], s);
    atomicAdd(&sumsq[c], q);
}

__global__ void k_bn_relu(float* __restrict__ h, const float* __restrict__ sum,
                          const float* __restrict__ sumsq, const float* __restrict__ g,
                          const float* __restrict__ be, long total, int C, float inv_n) {
    long t = (long)blockIdx.x * blockDim.x + threadIdx.x;
    if (t >= total) return;
    int c = (int)(t % C);
    float mu  = sum[c] * inv_n;
    float var = sumsq[c] * inv_n - mu * mu;
    float v = (h[t] - mu) * rsqrtf(var + 1e-5f) * g[c] + be[c];
    h[t] = v > 0.f ? v : 0.f;
}

// ---- SAGPool score: dot(agg, p_rel) + dot(x, p_root) + b  (wave per node) --
__global__ void k_score(const float* __restrict__ agg, const float* __restrict__ x,
                        const float* __restrict__ prel, const float* __restrict__ proot,
                        const float* __restrict__ pb, float* __restrict__ score,
                        int n, int C) {
    int wave = threadIdx.x >> 5, lane = threadIdx.x & 31;
    int i = blockIdx.x * 8 + wave;
    if (i >= n) return;
    float s = 0.f;
    for (int c = lane; c < C; c += 32)
        s += agg[(long)i * C + c] * prel[c] + x[(long)i * C + c] * proot[c];
    for (int m = 16; m >= 1; m >>= 1) s += __shfl_xor(s, m, 32);
    if (lane == 0) score[i] = s + pb[0];
}

// ---- per-graph top-k via 4-pass radix select (LDS histogram) ---------------
__global__ void k_topk_thresh(const float* __restrict__ score, int npg, int k,
                              unsigned* __restrict__ thr, int* __restrict__ cgt) {
    __shared__ int hist[256];
    __shared__ unsigned s_prefix, s_pmask;
    __shared__ int s_k, s_gt;
    const float* sc = score + (long)blockIdx.x * npg;
    if (threadIdx.x == 0) { s_prefix = 0u; s_pmask = 0u; s_k = k; s_gt = 0; }
    __syncthreads();
    for (int shift = 24; shift >= 0; shift -= 8) {
        hist[threadIdx.x] = 0;
        __syncthreads();
        unsigned pm = s_pmask, pf = s_prefix;
        for (int i = threadIdx.x; i < npg; i += blockDim.x) {
            unsigned key = f2ord(sc[i]);
            if ((key & pm) == pf) atomicAdd(&hist[(key >> shift) & 255], 1);
        }
        __syncthreads();
        if (threadIdx.x == 0) {
            int kk = s_k, gt = s_gt, b;
            for (b = 255; b > 0; --b) {
                int c = hist[b];
                if (kk <= c) break;
                kk -= c; gt += c;
            }
            s_k = kk; s_gt = gt;
            s_prefix |= (unsigned)b << shift;
            s_pmask  |= 0xFFu << shift;
        }
        __syncthreads();
    }
    if (threadIdx.x == 0) { thr[blockIdx.x] = s_prefix; cgt[blockIdx.x] = s_gt; }
}

__global__ void k_compact(const float* __restrict__ score, int npg, int k,
                          const unsigned* __restrict__ thr, const int* __restrict__ cgt,
                          int* __restrict__ counters, int* __restrict__ perm) {
    int g = blockIdx.x;
    unsigned T = thr[g];
    int cg = cgt[g];
    const float* sc = score + (long)g * npg;
    for (int i = threadIdx.x; i < npg; i += blockDim.x) {
        unsigned key = f2ord(sc[i]);
        if (key > T) {
            int p = atomicAdd(&counters[2 * g], 1);
            perm[(long)g * k + p] = g * npg + i;
        } else if (key == T) {
            int q = atomicAdd(&counters[2 * g + 1], 1);
            if (q < k - cg) perm[(long)g * k + cg + q] = g * npg + i;
        }
    }
}

__global__ void k_inv(const int* __restrict__ perm, int* __restrict__ inv, int m) {
    int j = blockIdx.x * blockDim.x + threadIdx.x;
    if (j < m) inv[perm[j]] = j;
}

__global__ void k_remap(const int* __restrict__ row, const int* __restrict__ col,
                        const float* __restrict__ w, const int* __restrict__ inv,
                        int* __restrict__ row2, int* __restrict__ col2,
                        float* __restrict__ w2, int E) {
    int e = blockIdx.x * blockDim.x + threadIdx.x;
    if (e >= E) return;
    int nr = inv[row[e]], nc = inv[col[e]];
    float wt = w[e];
    bool ok = (nr >= 0) && (nc >= 0) && (wt > 0.f);
    row2[e] = ok ? nr : 0;
    col2[e] = ok ? nc : 0;
    w2[e]   = ok ? wt : 0.f;
}

__global__ void k_gather(const float* __restrict__ h, const float* __restrict__ score,
                         const int* __restrict__ perm, float* __restrict__ xo,
                         int m, int C) {
    long t = (long)blockIdx.x * blockDim.x + threadIdx.x;
    int per = C >> 2;
    long j = t / per;
    if (j >= m) return;
    int c = (int)(t % per) << 2;
    int p = perm[j];
    float th = tanhf(score[p]);
    const float4 hv = *(const float4*)(h + (long)p * C + c);
    float4 o = { hv.x * th, hv.y * th, hv.z * th, hv.w * th };
    *(float4*)(xo + j * (long)C + c) = o;
}

// ---- readout: per-graph [max | mean] over nodes, then FC -------------------

__global__ void k_feature(const float* __restrict__ h, int kpg, int C,
                          float* __restrict__ feat) {
    int g = blockIdx.x, c = threadIdx.x;           // C == 256, block == 256
    const float* base = h + (long)g * kpg * C + c;
    float mx = -3.0e38f, sm = 0.f;
    for (int r = 0; r < kpg; ++r) {
        float v = base[(long)r * C];
        mx = fmaxf(mx, v); sm += v;
    }
    feat[(long)g * 2 * C + c]     = mx;
    feat[(long)g * 2 * C + C + c] = sm / (float)kpg;
}

__global__ void k_fc(const float* __restrict__ feat, const float* __restrict__ fcW,
                     const float* __restrict__ fcb, float* __restrict__ out) {
    int wave = threadIdx.x >> 5, lane = threadIdx.x & 31;   // 8 waves = 8 graphs
    float s = 0.f;
    for (int c = lane; c < 512; c += 32) s += feat[wave * 512 + c] * fcW[c];
    for (int m = 16; m >= 1; m >>= 1) s += __shfl_xor(s, m, 32);
    if (lane == 0) out[wave] = s + fcb[0];
}

// ---------------------------------------------------------------------------
// host-side drivers
// ---------------------------------------------------------------------------

static void gcn_layer(const float* X, int n, int Cin, int Cout,
                      const __bf16* Wt, const float* bias,
                      const float* g, const float* be,
                      const int* row, const int* col, const float* w, int E,
                      float* Hpre, float* Hout, float* deg, float* dinv,
                      float* colsum, float* colsumsq, bool bn_relu,
                      hipStream_t stream) {
    hipMemsetAsync(deg, 0, (size_t)n * sizeof(float), stream);
    k_deg<<<(E + 255) / 256, 256, 0, stream>>>(col, w, deg, E);
    k_dinv<<<(n + 255) / 256, 256, 0, stream>>>(deg, dinv, n);

    dim3 gg(n / 64, Cout / 64);
    size_t shbytes = (size_t)64 * Cin * sizeof(__bf16);   // B panel in LDS
    k_gemm_bf16<<<gg, 128, shbytes, stream>>>(X, Wt, Hpre, n, Cin, Cout);

    long total = (long)n * Cout;
    k_bias_init<<<(unsigned)((total + 255) / 256), 256, 0, stream>>>(Hout, bias, total, Cout);
    long st = (long)E * (Cout >> 2);
    k_scatter<<<(unsigned)((st + 255) / 256), 256, 0, stream>>>(Hpre, row, col, w, dinv, Hout, E, Cout);

    if (bn_relu) {
        hipMemsetAsync(colsum,   0, Cout * sizeof(float), stream);
        hipMemsetAsync(colsumsq, 0, Cout * sizeof(float), stream);
        int splits = 64;
        int rowsPer = (n + splits - 1) / splits;
        dim3 sg((Cout + 255) / 256, splits);
        k_colstats<<<sg, 256, 0, stream>>>(Hout, n, Cout, rowsPer, colsum, colsumsq);
        k_bn_relu<<<(unsigned)((total + 255) / 256), 256, 0, stream>>>(
            Hout, colsum, colsumsq, g, be, total, Cout, 1.0f / (float)n);
    }
}

static void sag_layer(const float* Hout, int n, int C, int npg, int k,
                      const float* prel, const float* proot, const float* pb,
                      const int* row, const int* col, const float* w, int E,
                      int* row2, int* col2, float* w2,
                      float* Agg, float* score, unsigned* thr, int* cgt,
                      int* counters, int* perm, int* inv, float* Xout,
                      hipStream_t stream) {
    hipMemsetAsync(Agg, 0, (size_t)n * C * sizeof(float), stream);
    long st = (long)E * (C >> 2);
    k_scatter<<<(unsigned)((st + 255) / 256), 256, 0, stream>>>(Hout, row, col, w,
                                                                nullptr, Agg, E, C);
    k_score<<<(n + 7) / 8, 256, 0, stream>>>(Agg, Hout, prel, proot, pb, score, n, C);
    k_topk_thresh<<<BGRAPH, 256, 0, stream>>>(score, npg, k, thr, cgt);
    hipMemsetAsync(counters, 0, 2 * BGRAPH * sizeof(int), stream);
    k_compact<<<BGRAPH, 256, 0, stream>>>(score, npg, k, thr, cgt, counters, perm);
    int m = BGRAPH * k;
    hipMemsetAsync(inv, 0xFF, (size_t)n * sizeof(int), stream);
    k_inv<<<(m + 255) / 256, 256, 0, stream>>>(perm, inv, m);
    k_remap<<<(E + 255) / 256, 256, 0, stream>>>(row, col, w, inv, row2, col2, w2, E);
    long gt = (long)m * (C >> 2);
    k_gather<<<(unsigned)((gt + 255) / 256), 256, 0, stream>>>(Hout, score, perm, Xout, m, C);
}

extern "C" void kernel_launch(void* const* d_in, const int* in_sizes, int n_in,
                              void* d_out, int out_size, void* d_ws, size_t ws_size,
                              hipStream_t stream) {
    (void)in_sizes; (void)n_in; (void)out_size; (void)ws_size;

    const float* x      = (const float*)d_in[0];
    const int*   ei     = (const int*)  d_in[1];
    const float* W1     = (const float*)d_in[2];
    const float* b1     = (const float*)d_in[3];
    const float* g1     = (const float*)d_in[4];
    const float* be1    = (const float*)d_in[5];
    const float* p1rel  = (const float*)d_in[6];
    const float* p1root = (const float*)d_in[7];
    const float* p1b    = (const float*)d_in[8];
    const float* W2     = (const float*)d_in[9];
    const float* b2     = (const float*)d_in[10];
    const float* g2     = (const float*)d_in[11];
    const float* be2    = (const float*)d_in[12];
    const float* p2rel  = (const float*)d_in[13];
    const float* p2root = (const float*)d_in[14];
    const float* p2b    = (const float*)d_in[15];
    const float* W3     = (const float*)d_in[16];
    const float* b3     = (const float*)d_in[17];
    const float* g3     = (const float*)d_in[18];
    const float* be3    = (const float*)d_in[19];
    const float* p3rel  = (const float*)d_in[20];
    const float* p3root = (const float*)d_in[21];
    const float* p3b    = (const float*)d_in[22];
    const float* W4     = (const float*)d_in[23];
    const float* b4     = (const float*)d_in[24];
    const float* fcW    = (const float*)d_in[25];
    const float* fcb    = (const float*)d_in[26];

    const int n0 = 48000, n1 = 28800, n2 = 17280, n3 = 8640;
    const int K1 = 3600, K2 = 2160, K3 = 1080;
    const int E = EDGES;

    // ---- workspace carving (~267 MB) ----
    char* p = (char*)d_ws;
    auto carve = [&](size_t bytes) {
        void* r = (void*)p;
        p += (bytes + 255) & ~(size_t)255;
        return r;
    };
    float*   Hpre     = (float*)carve((size_t)n0 * 512 * 4);   // also reused as Agg
    float*   Hout     = (float*)carve((size_t)n0 * 512 * 4);
    float*   Xcur     = (float*)carve((size_t)n1 * 512 * 4);
    __bf16*  Wt1      = (__bf16*)carve((size_t)512 * 512 * 2);
    __bf16*  Wt2      = (__bf16*)carve((size_t)512 * 512 * 2);
    __bf16*  Wt3      = (__bf16*)carve((size_t)256 * 512 * 2);
    __bf16*  Wt4      = (__bf16*)carve((size_t)256 * 256 * 2);
    int*     rowA     = (int*)  carve((size_t)E * 4);
    int*     colA     = (int*)  carve((size_t)E * 4);
    float*   wA       = (float*)carve((size_t)E * 4);
    int*     rowB     = (int*)  carve((size_t)E * 4);
    int*     colB     = (int*)  carve((size_t)E * 4);
    float*   wB       = (float*)carve((size_t)E * 4);
    float*   deg      = (float*)carve((size_t)n0 * 4);
    float*   dinv     = (float*)carve((size_t)n0 * 4);
    float*   score    = (float*)carve((size_t)n0 * 4);
    float*   colsum   = (float*)carve(512 * 4);
    float*   colsumsq = (float*)carve(512 * 4);
    int*     perm     = (int*)  carve((size_t)n1 * 4);
    int*     inv      = (int*)  carve((size_t)n0 * 4);
    unsigned* thr     = (unsigned*)carve(BGRAPH * 4);
    int*     cgt      = (int*)  carve(BGRAPH * 4);
    int*     counters = (int*)  carve(2 * BGRAPH * 4);

    // ---- weight conversion (bf16, transposed) + initial edge state ----
    k_wt_bf16<<<(512 * 512 + 255) / 256, 256, 0, stream>>>(W1, Wt1, 512, 512);
    k_wt_bf16<<<(512 * 512 + 255) / 256, 256, 0, stream>>>(W2, Wt2, 512, 512);
    k_wt_bf16<<<(512 * 256 + 255) / 256, 256, 0, stream>>>(W3, Wt3, 512, 256);
    k_wt_bf16<<<(256 * 256 + 255) / 256, 256, 0, stream>>>(W4, Wt4, 256, 256);
    k_init_edges<<<(E + 255) / 256, 256, 0, stream>>>(ei, rowA, colA, wA, E);

    // ---- layer 1 ----
    gcn_layer(x, n0, 512, 512, Wt1, b1, g1, be1, rowA, colA, wA, E,
              Hpre, Hout, deg, dinv, colsum, colsumsq, true, stream);
    sag_layer(Hout, n0, 512, 6000, K1, p1rel, p1root, p1b, rowA, colA, wA, E,
              rowB, colB, wB, Hpre, score, thr, cgt, counters, perm, inv, Xcur, stream);

    // ---- layer 2 ----
    gcn_layer(Xcur, n1, 512, 512, Wt2, b2, g2, be2, rowB, colB, wB, E,
              Hpre, Hout, deg, dinv, colsum, colsumsq, true, stream);
    sag_layer(Hout, n1, 512, 3600, K2, p2rel, p2root, p2b, rowB, colB, wB, E,
              rowA, colA, wA, Hpre, score, thr, cgt, counters, perm, inv, Xcur, stream);

    // ---- layer 3 (512 -> 256) ----
    gcn_layer(Xcur, n2, 512, 256, Wt3, b3, g3, be3, rowA, colA, wA, E,
              Hpre, Hout, deg, dinv, colsum, colsumsq, true, stream);
    sag_layer(Hout, n2, 256, 2160, K3, p3rel, p3root, p3b, rowA, colA, wA, E,
              rowB, colB, wB, Hpre, score, thr, cgt, counters, perm, inv, Xcur, stream);

    // ---- conv4 (no BN/ReLU, bias kept) ----
    gcn_layer(Xcur, n3, 256, 256, Wt4, b4, nullptr, nullptr, rowB, colB, wB, E,
              Hpre, Hout, deg, dinv, colsum, colsumsq, false, stream);

    // ---- readout: d_out = [out(8) | feature(8x512)] ----
    float* outp = (float*)d_out;
    float* feat = outp + 8;
    k_feature<<<BGRAPH, 256, 0, stream>>>(Hout, K3, 256, feat);
    k_fc<<<1, 256, 0, stream>>>(feat, fcW, fcb, outp);
}
